// RQV_9655086481438
// MI455X (gfx1250) — matrix operand order repaired and checked by reading, not compiled, not run on hardware
//
#include <hip/hip_runtime.h>

#define GAMMA    0.99f
#define D_DIM    256
#define S_DIM    2048
#define B_DIM    16
#define NCODE    1024
#define NQ       8
#define NROWS    (B_DIM * S_DIM)          // 32768

typedef __bf16 bf16_t;
typedef __attribute__((ext_vector_type(16))) __bf16 v16bf;
typedef __attribute__((ext_vector_type(8)))  __bf16 v8bf;
typedef __attribute__((ext_vector_type(8)))  float  v8f;

__device__ __forceinline__ unsigned short bf_bits(float f) {
    unsigned u = __builtin_bit_cast(unsigned, f);
    unsigned r = u + 0x7FFFu + ((u >> 16) & 1u);   // round-to-nearest-even
    return (unsigned short)(r >> 16);
}

// ---------------------------------------------------------------------------
// K0: residual[bs][d] = data[b][d][s]; also bf16 copy; zero loss slot.
// ---------------------------------------------------------------------------
__global__ __launch_bounds__(256) void rq_init(
    const float* __restrict__ data, float* __restrict__ residual,
    unsigned short* __restrict__ res_bf, float* __restrict__ loss_slot)
{
    const int bs = blockIdx.x;            // b*S + s
    const int d  = threadIdx.x;
    const int b  = bs >> 11;
    const int s  = bs & (S_DIM - 1);
    const float v = data[(size_t)b * D_DIM * S_DIM + (size_t)d * S_DIM + s];
    residual[(size_t)bs * D_DIM + d] = v;
    res_bf  [(size_t)bs * D_DIM + d] = bf_bits(v);
    if (bs == 0 && d == 0) *loss_slot = 0.0f;
}

// ---------------------------------------------------------------------------
// K1 (per stage): bf16 codebook + ||w||^2
// ---------------------------------------------------------------------------
__global__ __launch_bounds__(256) void rq_prep_codebook(
    const float* __restrict__ W, unsigned short* __restrict__ w_bf,
    float* __restrict__ wnorm)
{
    __shared__ float sred[256];
    const int code = blockIdx.x, d = threadIdx.x;
    const float w = W[(size_t)code * D_DIM + d];
    w_bf[(size_t)code * D_DIM + d] = bf_bits(w);
    sred[d] = w * w;
    __syncthreads();
    for (int t = 128; t > 0; t >>= 1) {
        if (d < t) sred[d] += sred[d + t];
        __syncthreads();
    }
    if (d == 0) wnorm[code] = sred[0];
}

// ---------------------------------------------------------------------------
// K2 (per stage): zero counts + vecsum
// ---------------------------------------------------------------------------
__global__ __launch_bounds__(256) void rq_zero(
    float* __restrict__ counts, float* __restrict__ vecsum)
{
    const int t = blockIdx.x * 256 + threadIdx.x;   // grid 1024*256 = 262144
    if (t < NCODE) counts[t] = 0.0f;
    vecsum[t] = 0.0f;
}

// ---------------------------------------------------------------------------
// K3 (per stage): WMMA distance GEMM fused with argmin.
// Block: 256 thr (8 waves) = one 16-row tile vs all 1024 codes.
// Wave w owns codes [w*128, w*128+128) = 8 col-tiles of 16.
// score[i][j] = ||w_j||^2 - 2 * x_i . w_j   (||x||^2 row-constant, dropped)
// A tile staged memory->LDS with GLOBAL_LOAD_ASYNC_TO_LDS_B128 (ASYNCcnt).
// ---------------------------------------------------------------------------
__global__ __launch_bounds__(256) void rq_argmin_wmma(
    const unsigned short* __restrict__ res_bf,   // [NROWS][256] bf16 bits
    const unsigned short* __restrict__ w_bf,     // [1024][256]  bf16 bits
    const float* __restrict__ wnorm,             // [1024]
    int*   __restrict__ idx_out,                 // [NROWS]
    float* __restrict__ idx_f_out)               // d_out indices region (stage slice)
{
    __shared__ unsigned short lds_a[16 * D_DIM]; // 8 KB A tile
    __shared__ float red_val[8 * 16];
    __shared__ int   red_idx[8 * 16];

    const int tid  = threadIdx.x;
    const int wave = tid >> 5;
    const int lane = tid & 31;
    const int row0 = blockIdx.x * 16;
    const int m    = lane & 15;
    const int half = lane >> 4;

    // Async-stage A tile (16 rows x 256 bf16 = 8 KB = 512 x 16B chunks).
    // Generic shared pointers carry the LDS byte offset in their low 32 bits.
    {
        const unsigned short* src = res_bf + (size_t)row0 * D_DIM;
#pragma unroll
        for (int j = 0; j < 2; ++j) {
            const int c = tid + j * 256;
            unsigned lds_addr = (unsigned)(size_t)(&lds_a[c * 8]);
            unsigned long long gaddr = (unsigned long long)(size_t)(src + c * 8);
            asm volatile("global_load_async_to_lds_b128 %0, %1, off"
                         :: "v"(lds_addr), "v"(gaddr) : "memory");
        }
        asm volatile("s_wait_asynccnt 0" ::: "memory");
    }
    __syncthreads();

    // Build 8 A fragments (one per K-tile of 32), contiguous K order, same as B:
    // lane (m, half): elems 0..7 -> K = t*32 + half*16 + e
    //                 elems 8..15 -> K = t*32 + half*16 + 8 + e
    v16bf afrag[8];
#pragma unroll
    for (int t = 0; t < 8; ++t) {
        const unsigned short* base = &lds_a[m * D_DIM + t * 32 + half * 16];
        v8bf lo = *(const v8bf*)(base);
        v8bf hi = *(const v8bf*)(base + 8);
#pragma unroll
        for (int e = 0; e < 8; ++e) { afrag[t][e] = lo[e]; afrag[t][8 + e] = hi[e]; }
    }

    float minv[8];
    int   mini[8];
#pragma unroll
    for (int r = 0; r < 8; ++r) { minv[r] = 3.4e38f; mini[r] = 0; }

    const int code_base = wave * 128;
    for (int ct = 0; ct < 8; ++ct) {
        const int col = code_base + ct * 16 + m;   // this lane's code column
        v8f c = {};
#pragma unroll
        for (int t = 0; t < 8; ++t) {
            // B (32x16): lane half owns contiguous K run of 16 for its column.
            const unsigned short* bbase =
                w_bf + (size_t)col * D_DIM + t * 32 + half * 16;
            v16bf bfrag = *(const v16bf*)(bbase);
            c = __builtin_amdgcn_wmma_f32_16x16x32_bf16(
                    false, afrag[t], false, bfrag, (short)0, c, false, false);
        }
        const float wn = wnorm[col];
#pragma unroll
        for (int r = 0; r < 8; ++r) {
            const float s = wn - 2.0f * c[r];
            if (s < minv[r]) { minv[r] = s; mini[r] = col; }
        }
    }

    // Cross-lane argmin within each 16-lane half (C layout: col = lane&15,
    // rows r (lanes 0-15) / 8+r (lanes 16-31) live in VGPR slot r).
#pragma unroll
    for (int r = 0; r < 8; ++r) {
        float v = minv[r]; int ix = mini[r];
#pragma unroll
        for (int off = 1; off < 16; off <<= 1) {
            const float ov = __shfl_xor(v, off, 32);
            const int   oi = __shfl_xor(ix, off, 32);
            if (ov < v || (ov == v && oi < ix)) { v = ov; ix = oi; }
        }
        if (m == 0) {
            const int row = half * 8 + r;
            red_val[wave * 16 + row] = v;
            red_idx[wave * 16 + row] = ix;
        }
    }
    __syncthreads();

    if (tid < 16) {
        float bv = red_val[tid]; int bi = red_idx[tid];
        for (int wv = 1; wv < 8; ++wv) {
            const float v = red_val[wv * 16 + tid];
            const int  ix = red_idx[wv * 16 + tid];
            if (v < bv || (v == bv && ix < bi)) { bv = v; bi = ix; }
        }
        idx_out  [row0 + tid] = bi;
        idx_f_out[row0 + tid] = (float)bi;
    }
}

// ---------------------------------------------------------------------------
// K4 (per stage): segment sums (pre-update residual), residual -= Q, bf16.
// Block = one sample; threads = D.
// ---------------------------------------------------------------------------
__global__ __launch_bounds__(256) void rq_update(
    float* __restrict__ residual, unsigned short* __restrict__ res_bf,
    const float* __restrict__ W, const int* __restrict__ idx,
    float* __restrict__ vecsum, float* __restrict__ counts)
{
    const int i = blockIdx.x;
    const int d = threadIdx.x;
    const int code = idx[i];
    const float r = residual[(size_t)i * D_DIM + d];
    atomicAdd(&vecsum[(size_t)code * D_DIM + d], r);       // flat BEFORE subtract
    const float nr = r - W[(size_t)code * D_DIM + d];
    residual[(size_t)i * D_DIM + d] = nr;
    res_bf  [(size_t)i * D_DIM + d] = bf_bits(nr);
    if (d == 0) atomicAdd(&counts[code], 1.0f);
}

// ---------------------------------------------------------------------------
// K5 (per stage): EMA outputs new_N / new_m / new_W
// ---------------------------------------------------------------------------
__global__ __launch_bounds__(256) void rq_finalize(
    const float* __restrict__ Ni, const float* __restrict__ mi,
    const float* __restrict__ counts, const float* __restrict__ vecsum,
    float* __restrict__ outN, float* __restrict__ outM, float* __restrict__ outW)
{
    const int code = blockIdx.x, d = threadIdx.x;
    const float nN = Ni[code] * GAMMA + counts[code] * (1.0f - GAMMA);
    if (d == 0) outN[code] = nN;
    const float nm = mi[(size_t)code * D_DIM + d] * GAMMA
                   + vecsum[(size_t)code * D_DIM + d] * (1.0f - GAMMA);
    outM[(size_t)code * D_DIM + d] = nm;
    outW[(size_t)code * D_DIM + d] = nm / fmaxf(nN, 1e-8f);
}

// ---------------------------------------------------------------------------
// K6: logits = data - residual (back to [B,D,S]); commitment loss = mean(res^2)
// ---------------------------------------------------------------------------
__global__ __launch_bounds__(256) void rq_final_out(
    const float* __restrict__ data, const float* __restrict__ residual,
    float* __restrict__ logits, float* __restrict__ loss)
{
    __shared__ float sred[256];
    const int bd = blockIdx.x;            // b*256 + d, grid = 4096
    const int b = bd >> 8, d = bd & 255;
    float acc = 0.0f;
    for (int s0 = 0; s0 < S_DIM; s0 += 256) {
        const int s = s0 + threadIdx.x;
        const size_t o = (size_t)b * D_DIM * S_DIM + (size_t)d * S_DIM + s;
        const float r = residual[((size_t)(b * S_DIM + s)) * D_DIM + d];
        logits[o] = data[o] - r;
        acc += r * r;
    }
    sred[threadIdx.x] = acc;
    __syncthreads();
    for (int t = 128; t > 0; t >>= 1) {
        if (threadIdx.x < t) sred[threadIdx.x] += sred[threadIdx.x + t];
        __syncthreads();
    }
    if (threadIdx.x == 0)
        atomicAdd(loss, sred[0] / (float)((size_t)B_DIM * S_DIM * D_DIM));
}

// ---------------------------------------------------------------------------
extern "C" void kernel_launch(void* const* d_in, const int* in_sizes, int n_in,
                              void* d_out, int out_size, void* d_ws, size_t ws_size,
                              hipStream_t stream)
{
    const float* data      = (const float*)d_in[0];   // [16,256,2048]
    const float* codebooks = (const float*)d_in[1];   // [8,1024,256]
    const float* N_i       = (const float*)d_in[2];   // [8,1024]
    const float* m_i       = (const float*)d_in[3];   // [8,1024,256]
    float* out = (float*)d_out;

    // Workspace layout (bytes)
    char* ws = (char*)d_ws;
    float*          residual = (float*)         (ws + 0);          // 32 MB
    unsigned short* res_bf   = (unsigned short*)(ws + 33554432);   // 16 MB
    unsigned short* w_bf     = (unsigned short*)(ws + 50331648);   // 512 KB
    float*          wnorm    = (float*)         (ws + 50855936);   // 4 KB
    float*          counts   = (float*)         (ws + 50860032);   // 4 KB
    float*          vecsum   = (float*)         (ws + 50864128);   // 1 MB
    int*            idx_ws   = (int*)           (ws + 51912704);   // 128 KB

    // Output layout (floats, return order)
    float* out_logits = out;                          // 16*256*2048
    float* out_loss   = out + 8388608;                // 1
    float* out_idx    = out + 8388609;                // 8*16*2048
    float* out_N      = out + 8650753;                // 8*1024
    float* out_m      = out + 8658945;                // 8*1024*256
    float* out_W      = out + 10756097;               // 8*1024*256

    rq_init<<<NROWS, 256, 0, stream>>>(data, residual, res_bf, out_loss);

    for (int q = 0; q < NQ; ++q) {
        const float* Wq = codebooks + (size_t)q * NCODE * D_DIM;
        rq_prep_codebook<<<NCODE, 256, 0, stream>>>(Wq, w_bf, wnorm);
        rq_zero<<<1024, 256, 0, stream>>>(counts, vecsum);
        rq_argmin_wmma<<<NROWS / 16, 256, 0, stream>>>(
            res_bf, w_bf, wnorm, idx_ws, out_idx + (size_t)q * NROWS);
        rq_update<<<NROWS, 256, 0, stream>>>(residual, res_bf, Wq, idx_ws,
                                             vecsum, counts);
        rq_finalize<<<NCODE, 256, 0, stream>>>(
            N_i + (size_t)q * NCODE, m_i + (size_t)q * NCODE * D_DIM,
            counts, vecsum,
            out_N + (size_t)q * NCODE,
            out_m + (size_t)q * NCODE * D_DIM,
            out_W + (size_t)q * NCODE * D_DIM);
    }

    rq_final_out<<<B_DIM * D_DIM, 256, 0, stream>>>(data, residual,
                                                    out_logits, out_loss);
}
